// HydraLAM_70205535420784
// MI455X (gfx1250) — matrix operand order repaired and checked by reading, not compiled, not run on hardware
//
#include <hip/hip_runtime.h>
#include <hip/hip_bf16.h>

typedef __attribute__((ext_vector_type(16))) _Float16 v16h;
typedef __attribute__((ext_vector_type(8)))  float    v8f;

#define DIM 384
#define SEQ 4096
#define NB  8
#define MTOT (NB*SEQ)          // 32768 rows
#define EPS 1e-5f

#define WMMA_F16(a, b, c) \
  __builtin_amdgcn_wmma_f32_16x16x32_f16(false, (a), false, (b), (short)0, (c), false, false)

// ---------------------------------------------------------------------------
// Packed fragment layouts (CDNA5 WMMA 16-bit operand layouts, ISA 7.12.2)
// A (MxK, f16): tile (tm,tk) of 16x32; lane = (m%16) | (hi<<4),
//   kk=k%32, hi=(kk>>3)&1, h=(kk&7)|((kk>>4)<<3); 16 halves/lane contiguous.
// B (KxN, f16): tile (tk,tn) of 32x16; lane = (n%16) | (hi<<4), same hi/h as A.
// ---------------------------------------------------------------------------
__device__ __forceinline__ size_t pa_index(int m, int k, int nTK) {
  int tm = m >> 4, r = m & 15;
  int tk = k >> 5, kk = k & 31;
  int hi = (kk >> 3) & 1;
  int h  = (kk & 7) | ((kk >> 4) << 3);
  int lane = r | (hi << 4);
  return ((((size_t)tm * nTK + tk) * 32 + lane) << 4) + h;
}
__device__ __forceinline__ size_t pb_index(int k, int n, int nTN) {
  int tk = k >> 5, kk = k & 31;
  int tn = n >> 4, nl = n & 15;
  int lane = nl | (((kk >> 3) & 1) << 4);
  int h = (kk & 7) | ((kk >> 4) << 3);
  return ((((size_t)tk * nTN + tn) * 32 + lane) << 4) + h;
}

// ---------------------------------------------------------------------------
// Pack one f32 row-major weight (K x Nsub) into packed-B f16 at column n0.
// ---------------------------------------------------------------------------
__global__ void k_pack_b(const float* __restrict__ src, _Float16* __restrict__ dst,
                         int K, int Ntot, int n0, int Nsub) {
  int idx = blockIdx.x * blockDim.x + threadIdx.x;
  if (idx >= K * Nsub) return;
  int k = idx / Nsub, nl = idx - k * Nsub;
  dst[pb_index(k, n0 + nl, Ntot >> 4)] = (_Float16)src[(size_t)k * Nsub + nl];
}

// ---------------------------------------------------------------------------
// Grouped conv1d (k=7, pad=3, groups=6) + SiLU -> packed-A f16 (M x 384)
// block: 384 threads (one out-channel each), 16 seq rows per block.
// All 32 lanes of a wave share one group -> LDS reads are broadcasts.
// ---------------------------------------------------------------------------
__global__ __launch_bounds__(384) void k_conv(const float* __restrict__ x,
                                              const float* __restrict__ cw,
                                              const float* __restrict__ cb,
                                              _Float16* __restrict__ xpa) {
  __shared__ float xt[22][DIM];
  int blk = blockIdx.x;
  int b   = blk >> 8;           // 8 batches
  int s0  = (blk & 255) * 16;   // 256 tiles of 16
  int o   = threadIdx.x;        // out channel
  const float* xb = x + (size_t)b * SEQ * DIM;
  for (int r = 0; r < 22; ++r) {
    int s = s0 - 3 + r;
    xt[r][o] = (s >= 0 && s < SEQ) ? xb[(size_t)s * DIM + o] : 0.f;
  }
  __syncthreads();
  float acc[16];
  float bias = cb[o];
#pragma unroll
  for (int si = 0; si < 16; ++si) acc[si] = bias;
  int ibase = (o >> 6) << 6;                 // group * 64
  const float* w = cw + (size_t)o * 64 * 7;
  for (int i = 0; i < 64; ++i) {
#pragma unroll
    for (int kk = 0; kk < 7; ++kk) {
      float wv = w[i * 7 + kk];
#pragma unroll
      for (int si = 0; si < 16; ++si)
        acc[si] = fmaf(xt[si + kk][ibase + i], wv, acc[si]);
    }
  }
#pragma unroll
  for (int si = 0; si < 16; ++si) {
    float v = acc[si];
    v = v / (1.f + __expf(-v));              // SiLU
    xpa[pa_index(b * SEQ + s0 + si, o, DIM / 32)] = (_Float16)v;
  }
}

// ---------------------------------------------------------------------------
// Fused QKVG GEMM: C(32768x1536) = Xc_f16 @ [q|k|v|g]_f16, WMMA f32 accum.
// grid (1024, 3): 32-row M tile x 128-col N group; 8 waves; each wave owns
// one 16-col n-tile replicated across the 4 weight blocks and both M tiles:
// 8 WMMAs per K-step fed by 6 fragment loads, double-buffered.
// Epilogue u = v*sigmoid(g)*k stays in-register.
// ---------------------------------------------------------------------------
__global__ __launch_bounds__(256) void k_gemm_qkvg(
    const _Float16* __restrict__ A, const _Float16* __restrict__ B,
    const float* __restrict__ qb, const float* __restrict__ kb,
    const float* __restrict__ vb, const float* __restrict__ gb,
    float* __restrict__ qout, float* __restrict__ uout) {
  const int tm2 = blockIdx.x;              // 32-row tile
  const int ng  = blockIdx.y;              // 0..2
  const int lane = threadIdx.x & 31;
  const int wave = threadIdx.x >> 5;
  const int nTK = DIM / 32;                // 12
  const int nTN = 1536 / 16;               // 96
  const int tnb = ng * 8 + wave;           // n-tile within each part, 0..23
  int tn[4];
#pragma unroll
  for (int p = 0; p < 4; ++p) tn[p] = p * 24 + tnb;

  v8f acc[4][2];
  v8f z = {};
#pragma unroll
  for (int p = 0; p < 4; ++p) { acc[p][0] = z; acc[p][1] = z; }

  const _Float16* Ab0 = A + (size_t)(tm2 * 2)     * nTK * 512 + lane * 16;
  const _Float16* Ab1 = A + (size_t)(tm2 * 2 + 1) * nTK * 512 + lane * 16;
  v16h a0[2], a1[2], b0[4], b1[4];
  a0[0] = *(const v16h*)(Ab0);
  a0[1] = *(const v16h*)(Ab1);
#pragma unroll
  for (int p = 0; p < 4; ++p)
    b0[p] = *(const v16h*)(B + (((size_t)0 * nTN + tn[p]) * 32 + lane) * 16);

#pragma unroll
  for (int kt = 0; kt < 12; ++kt) {
    int ktn = (kt + 1 < 12) ? kt + 1 : 11;
    a1[0] = *(const v16h*)(Ab0 + (size_t)ktn * 512);
    a1[1] = *(const v16h*)(Ab1 + (size_t)ktn * 512);
#pragma unroll
    for (int p = 0; p < 4; ++p)
      b1[p] = *(const v16h*)(B + (((size_t)ktn * nTN + tn[p]) * 32 + lane) * 16);
#pragma unroll
    for (int p = 0; p < 4; ++p) {
      acc[p][0] = WMMA_F16(a0[0], b0[p], acc[p][0]);
      acc[p][1] = WMMA_F16(a0[1], b0[p], acc[p][1]);
    }
    a0[0] = a1[0]; a0[1] = a1[1];
#pragma unroll
    for (int p = 0; p < 4; ++p) b0[p] = b1[p];
  }

  // epilogue: q out, u = v*sigmoid(g)*k out (row-major f32)
  int nl = lane & 15, mh = lane >> 4;
  int n = tnb * 16 + nl;                   // 0..383
  float bq = qb[n], bk = kb[n], bvv = vb[n], bg = gb[n];
#pragma unroll
  for (int mi = 0; mi < 2; ++mi) {
#pragma unroll
    for (int r = 0; r < 8; ++r) {
      int m = (tm2 * 2 + mi) * 16 + r + 8 * mh;
      float qv = acc[0][mi][r] + bq;
      float kv = acc[1][mi][r] + bk;
      float vv = acc[2][mi][r] + bvv;
      float gv = acc[3][mi][r] + bg;
      float sg = 1.f / (1.f + __expf(-gv));
      size_t off = (size_t)m * DIM + n;
      qout[off] = qv;
      uout[off] = vv * sg * kv;
    }
  }
}

// ---------------------------------------------------------------------------
// Chunked decayed scan + LN1 -> packed-A f16 h.
// lam^128 ~ 4e-20 => 128-step warm-up makes chunks independent in fp32.
// Reductions: wave32 shfl + 12-slot LDS combine (2 barriers per reduction).
// ---------------------------------------------------------------------------
__device__ __forceinline__ float blk_sum384(float v, float* part, int tid) {
  int lane = tid & 31, wid = tid >> 5;
#pragma unroll
  for (int o = 16; o > 0; o >>= 1) v += __shfl_xor(v, o, 32);
  __syncthreads();                 // protect previous reads of part[]
  if (lane == 0) part[wid] = v;
  __syncthreads();
  float s = 0.f;
#pragma unroll
  for (int i = 0; i < 12; ++i) s += part[i];
  return s;
}

__global__ __launch_bounds__(384) void k_scan(
    const float* __restrict__ qbuf, const float* __restrict__ ubuf,
    const float* __restrict__ decay, const float* __restrict__ g1,
    const float* __restrict__ b1, _Float16* __restrict__ hpa) {
  __shared__ float part[12];
  int b  = blockIdx.x >> 5;       // 8 batches
  int ch = blockIdx.x & 31;       // 32 chunks of 128
  int d  = threadIdx.x;
  float lam = blk_sum384(1.f / (1.f + __expf(-decay[d])), part, d) * (1.f / DIM);

  int s_begin = ch * 128;
  int warm = s_begin < 128 ? s_begin : 128;
  size_t base = (size_t)b * SEQ * DIM;
  float carry = 0.f;
  for (int s = s_begin - warm; s < s_begin; ++s)
    carry = carry * lam + ubuf[base + (size_t)s * DIM + d];
  float gg = g1[d], bb = b1[d];
  for (int s = s_begin; s < s_begin + 128; ++s) {
    size_t off = base + (size_t)s * DIM + d;
    carry = carry * lam + ubuf[off];
    float xm = qbuf[off] * carry;
    float mean = blk_sum384(xm, part, d) * (1.f / DIM);
    float dx = xm - mean;
    float var = blk_sum384(dx * dx, part, d) * (1.f / DIM);
    float hv = dx * rsqrtf(var + EPS) * gg + bb;
    hpa[pa_index(b * SEQ + s, d, DIM / 32)] = (_Float16)hv;
  }
}

// ---------------------------------------------------------------------------
// MLP GEMM1: h(32768x384) @ w1(384x768) + b1, SiLU -> packed-A f16 (K=768)
// grid (1024, 2); 3 n-tiles x 2 m-tiles per wave; double-buffered K loop.
// ---------------------------------------------------------------------------
__global__ __launch_bounds__(256) void k_mlp1(const _Float16* __restrict__ A,
                                              const _Float16* __restrict__ B,
                                              const float* __restrict__ bias,
                                              _Float16* __restrict__ a1pa) {
  const int tm2 = blockIdx.x;
  const int ng  = blockIdx.y;     // 0..1
  const int lane = threadIdx.x & 31;
  const int wave = threadIdx.x >> 5;
  const int nTK = DIM / 32;       // 12
  const int nTN = 768 / 16;       // 48
  int tn[3];
#pragma unroll
  for (int j = 0; j < 3; ++j) tn[j] = ng * 24 + wave * 3 + j;

  v8f acc[3][2];
  v8f z = {};
#pragma unroll
  for (int j = 0; j < 3; ++j) { acc[j][0] = z; acc[j][1] = z; }

  const _Float16* Ab0 = A + (size_t)(tm2 * 2)     * nTK * 512 + lane * 16;
  const _Float16* Ab1 = A + (size_t)(tm2 * 2 + 1) * nTK * 512 + lane * 16;
  v16h a0[2], a1[2], b0[3], b1[3];
  a0[0] = *(const v16h*)(Ab0);
  a0[1] = *(const v16h*)(Ab1);
#pragma unroll
  for (int j = 0; j < 3; ++j)
    b0[j] = *(const v16h*)(B + (((size_t)0 * nTN + tn[j]) * 32 + lane) * 16);

#pragma unroll
  for (int kt = 0; kt < 12; ++kt) {
    int ktn = (kt + 1 < 12) ? kt + 1 : 11;
    a1[0] = *(const v16h*)(Ab0 + (size_t)ktn * 512);
    a1[1] = *(const v16h*)(Ab1 + (size_t)ktn * 512);
#pragma unroll
    for (int j = 0; j < 3; ++j)
      b1[j] = *(const v16h*)(B + (((size_t)ktn * nTN + tn[j]) * 32 + lane) * 16);
#pragma unroll
    for (int j = 0; j < 3; ++j) {
      acc[j][0] = WMMA_F16(a0[0], b0[j], acc[j][0]);
      acc[j][1] = WMMA_F16(a0[1], b0[j], acc[j][1]);
    }
    a0[0] = a1[0]; a0[1] = a1[1];
#pragma unroll
    for (int j = 0; j < 3; ++j) b0[j] = b1[j];
  }

  int nl = lane & 15, mh = lane >> 4;
#pragma unroll
  for (int j = 0; j < 3; ++j) {
    int n = tn[j] * 16 + nl;
    float bv = bias[n];
#pragma unroll
    for (int mi = 0; mi < 2; ++mi) {
#pragma unroll
      for (int r = 0; r < 8; ++r) {
        int m = (tm2 * 2 + mi) * 16 + r + 8 * mh;
        float v = acc[j][mi][r] + bv;
        v = v / (1.f + __expf(-v));  // SiLU
        a1pa[pa_index(m, n, 768 / 32)] = (_Float16)v;
      }
    }
  }
}

// ---------------------------------------------------------------------------
// MLP GEMM2: a1(32768x768) @ w2(768x384) + b2 + x residual, then LN2 -> out
// 32-row M tile; LN over full 384-col row via 48KB LDS tile + wave32 shfl.
// Double-buffered K loop (24 steps).
// ---------------------------------------------------------------------------
__global__ __launch_bounds__(256) void k_mlp2(
    const _Float16* __restrict__ A, const _Float16* __restrict__ B,
    const float* __restrict__ b2, const float* __restrict__ xres,
    const float* __restrict__ g2, const float* __restrict__ bt2,
    float* __restrict__ out) {
  __shared__ float tile[32][DIM];
  const int tm2 = blockIdx.x;
  const int lane = threadIdx.x & 31;
  const int wave = threadIdx.x >> 5;
  const int nTK = 768 / 32;       // 24
  const int nTN = DIM / 16;       // 24
  int tn[3];
#pragma unroll
  for (int j = 0; j < 3; ++j) tn[j] = wave * 3 + j;

  v8f acc[3][2];
  v8f z = {};
#pragma unroll
  for (int j = 0; j < 3; ++j) { acc[j][0] = z; acc[j][1] = z; }

  const _Float16* Ab0 = A + (size_t)(tm2 * 2)     * nTK * 512 + lane * 16;
  const _Float16* Ab1 = A + (size_t)(tm2 * 2 + 1) * nTK * 512 + lane * 16;
  v16h a0[2], a1[2], b0[3], b1[3];
  a0[0] = *(const v16h*)(Ab0);
  a0[1] = *(const v16h*)(Ab1);
#pragma unroll
  for (int j = 0; j < 3; ++j)
    b0[j] = *(const v16h*)(B + (((size_t)0 * nTN + tn[j]) * 32 + lane) * 16);

#pragma unroll
  for (int kt = 0; kt < 24; ++kt) {
    int ktn = (kt + 1 < 24) ? kt + 1 : 23;
    a1[0] = *(const v16h*)(Ab0 + (size_t)ktn * 512);
    a1[1] = *(const v16h*)(Ab1 + (size_t)ktn * 512);
#pragma unroll
    for (int j = 0; j < 3; ++j)
      b1[j] = *(const v16h*)(B + (((size_t)ktn * nTN + tn[j]) * 32 + lane) * 16);
#pragma unroll
    for (int j = 0; j < 3; ++j) {
      acc[j][0] = WMMA_F16(a0[0], b0[j], acc[j][0]);
      acc[j][1] = WMMA_F16(a0[1], b0[j], acc[j][1]);
    }
    a0[0] = a1[0]; a0[1] = a1[1];
#pragma unroll
    for (int j = 0; j < 3; ++j) b0[j] = b1[j];
  }

  // stage (gemm + bias + residual) into LDS row tile
  int nl = lane & 15, mh = lane >> 4;
#pragma unroll
  for (int j = 0; j < 3; ++j) {
    int n = tn[j] * 16 + nl;
    float bv = b2[n];
#pragma unroll
    for (int mi = 0; mi < 2; ++mi) {
#pragma unroll
      for (int r = 0; r < 8; ++r) {
        int row = mi * 16 + r + 8 * mh;
        int m = tm2 * 32 + row;
        tile[row][n] = acc[j][mi][r] + bv + xres[(size_t)m * DIM + n];
      }
    }
  }
  __syncthreads();
  // LN2: each wave handles 4 rows; 12 cols per lane
#pragma unroll
  for (int rr = 0; rr < 4; ++rr) {
    int row = wave * 4 + rr;
    float s = 0.f, sq = 0.f;
#pragma unroll
    for (int t = 0; t < 12; ++t) {
      float v = tile[row][lane + 32 * t];
      s += v; sq += v * v;
    }
#pragma unroll
    for (int o = 16; o > 0; o >>= 1) {
      s  += __shfl_xor(s, o, 32);
      sq += __shfl_xor(sq, o, 32);
    }
    float mean = s * (1.f / DIM);
    float var = sq * (1.f / DIM) - mean * mean;
    float inv = rsqrtf(var + EPS);
    size_t mbase = (size_t)(tm2 * 32 + row) * DIM;
#pragma unroll
    for (int t = 0; t < 12; ++t) {
      int c = lane + 32 * t;
      out[mbase + c] = (tile[row][c] - mean) * inv * g2[c] + bt2[c];
    }
  }
}

// ---------------------------------------------------------------------------
extern "C" void kernel_launch(void* const* d_in, const int* in_sizes, int n_in,
                              void* d_out, int out_size, void* d_ws, size_t ws_size,
                              hipStream_t stream) {
  const float* x      = (const float*)d_in[0];
  const float* conv_w = (const float*)d_in[1];
  const float* conv_b = (const float*)d_in[2];
  const float* q_w    = (const float*)d_in[3];
  const float* q_b    = (const float*)d_in[4];
  const float* k_w    = (const float*)d_in[5];
  const float* k_b    = (const float*)d_in[6];
  const float* v_w    = (const float*)d_in[7];
  const float* v_b    = (const float*)d_in[8];
  const float* g_w    = (const float*)d_in[9];
  const float* g_b    = (const float*)d_in[10];
  const float* decay  = (const float*)d_in[11];
  const float* ln1_g  = (const float*)d_in[12];
  const float* ln1_b  = (const float*)d_in[13];
  const float* w1     = (const float*)d_in[14];
  const float* b1     = (const float*)d_in[15];
  const float* w2     = (const float*)d_in[16];
  const float* b2     = (const float*)d_in[17];
  const float* ln2_g  = (const float*)d_in[18];
  const float* ln2_b  = (const float*)d_in[19];
  float* out = (float*)d_out;

  char* ws = (char*)d_ws;
  _Float16* wqkvg = (_Float16*)(ws + 0);           // 384x1536 f16 packed-B
  _Float16* w1p   = (_Float16*)(ws + 1179648);     // 384x768
  _Float16* w2p   = (_Float16*)(ws + 1769472);     // 768x384
  _Float16* xcp   = (_Float16*)(ws + 2359296);     // 32768x384 packed-A
  float*    qbuf  = (float*)   (ws + 27525120);    // 32768x384 f32
  float*    ubuf  = (float*)   (ws + 77856768);    // 32768x384 f32
  _Float16* hpa   = (_Float16*)(ws + 128188416);   // 32768x384 packed-A
  _Float16* a1p   = (_Float16*)(ws + 153354240);   // 32768x768 packed-A
  // end: 203685888 bytes

  // pack weights to WMMA fragment layout
  int t384 = (DIM * DIM + 255) / 256;
  k_pack_b<<<t384, 256, 0, stream>>>(q_w, wqkvg, DIM, 1536, 0,    DIM);
  k_pack_b<<<t384, 256, 0, stream>>>(k_w, wqkvg, DIM, 1536, 384,  DIM);
  k_pack_b<<<t384, 256, 0, stream>>>(v_w, wqkvg, DIM, 1536, 768,  DIM);
  k_pack_b<<<t384, 256, 0, stream>>>(g_w, wqkvg, DIM, 1536, 1152, DIM);
  int t768 = (DIM * 768 + 255) / 256;
  k_pack_b<<<t768, 256, 0, stream>>>(w1, w1p, DIM, 768, 0, 768);
  k_pack_b<<<t768, 256, 0, stream>>>(w2, w2p, 768, DIM, 0, DIM);

  k_conv<<<NB * 256, 384, 0, stream>>>(x, conv_w, conv_b, xcp);
  k_gemm_qkvg<<<dim3(MTOT / 32, 3), 256, 0, stream>>>(xcp, wqkvg, q_b, k_b, v_b, g_b,
                                                      qbuf, ubuf);
  k_scan<<<NB * 32, 384, 0, stream>>>(qbuf, ubuf, decay, ln1_g, ln1_b, hpa);
  k_mlp1<<<dim3(MTOT / 32, 2), 256, 0, stream>>>(hpa, w1p, b1, a1p);
  k_mlp2<<<MTOT / 32, 256, 0, stream>>>(a1p, w2p, b2, x, ln2_g, ln2_b, out);
}
